// Decoder_26852135535210
// MI455X (gfx1250) — compile-verified, matrix operand order checked by reading, not run on hardware
//
#include <hip/hip_runtime.h>
#include <stdint.h>

// ---------------------------------------------------------------------------
// Attention-LSTM decoder for MI455X (gfx1250, wave32, WMMA + TDM).
// One-time pass converts all GEMM operands to bf16 workspace buffers; hot
// loops are pure b128 loads + v_wmma_f32_16x16x32_bf16. W_ih tiles are staged
// into LDS by the Tensor Data Mover (6-arg builtin form), h-vector staged by
// async global->LDS.
// ---------------------------------------------------------------------------

#define B_    64
#define S_    256
#define E_    512
#define H_    1024
#define D2H   2048   // 2*H
#define NG    4096   // 4*H
#define KI    1536   // E+H
#define KC    3072   // 3*H
#define TSTEPS 63
#define KCHUNK 512   // TDM K-chunk for W_ih staging

typedef __attribute__((ext_vector_type(16))) __bf16          v16bf;
typedef __attribute__((ext_vector_type(8)))  float           v8f;
typedef __attribute__((ext_vector_type(16))) unsigned short  v16us;
typedef __attribute__((ext_vector_type(4)))  unsigned int    v4u;
typedef __attribute__((ext_vector_type(4)))  unsigned int    u32x4;
typedef __attribute__((ext_vector_type(8)))  int             i32x8;
typedef __attribute__((ext_vector_type(4)))  int             i32x4;
typedef __attribute__((ext_vector_type(2)))  int             v2i;
typedef __attribute__((address_space(3)))    v2i             lds_v2i;

#if __has_builtin(__builtin_amdgcn_tensor_load_to_lds)
#define USE_TDM 1
#endif

union TileU { v16us s; v16bf b; v4u q[2]; };

__device__ __forceinline__ unsigned short f2bf(float f) {
    union { __bf16 h; unsigned short s; } u;
    u.h = (__bf16)f;                       // native v_cvt on gfx1250
    return u.s;
}
__device__ __forceinline__ float bf2f(unsigned short b) {
    union { unsigned short s; __bf16 h; } u; u.s = b;
    return (float)u.h;
}
__device__ __forceinline__ float sigf(float x) { return 1.0f / (1.0f + __expf(-x)); }

__device__ __forceinline__ v8f wmma_bf16(v16bf a, v16bf b, v8f c) {
    return __builtin_amdgcn_wmma_f32_16x16x32_bf16(false, a, false, b, (short)0, c,
                                                   false, false);
}

// 16x32 bf16 tile from row-major bf16 memory (global or LDS), wave32 layout:
//   lane<16 : row = lane,    K runs {k0+0..7},  {k0+16..23}
//   lane>=16: row = lane-16, K runs {k0+8..15}, {k0+24..31}
__device__ __forceinline__ v16bf tile_bf(const unsigned short* base, int ld, int k0) {
    const int lane = threadIdx.x & 31;
    const int r    = lane & 15;
    const int kofs = (lane >> 4) << 3;
    const unsigned short* p = base + (size_t)r * ld + k0 + kofs;
    TileU t;
    t.q[0] = *(const v4u*)(p);
    t.q[1] = *(const v4u*)(p + 16);
    return t.b;
}

// ---------------------------------------------------------------------------
// One-time f32 -> bf16 conversion (grid-stride).
// ---------------------------------------------------------------------------
__global__ void k_cvt(const float* __restrict__ src, unsigned short* __restrict__ dst,
                      int n) {
    for (int i = blockIdx.x * blockDim.x + threadIdx.x; i < n;
         i += gridDim.x * blockDim.x)
        dst[i] = f2bf(src[i]);
}

__global__ void k_zero(v4u* __restrict__ p, int n4) {
    const int i = blockIdx.x * blockDim.x + threadIdx.x;
    if (i < n4) { v4u z = {0u, 0u, 0u, 0u}; p[i] = z; }
}

// y_bf[t,b,k] = bf16(embed[tgt[t,b], k]) for t in [0,63)
__global__ void k_embed(const int* __restrict__ tgt, const float* __restrict__ embed,
                        unsigned short* __restrict__ y_bf) {
    const int n = TSTEPS * B_ * E_;
    for (int i = blockIdx.x * blockDim.x + threadIdx.x; i < n;
         i += gridDim.x * blockDim.x) {
        const int row = i >> 9;            // t*B + b
        const int k   = i & (E_ - 1);
        y_bf[i] = f2bf(embed[(size_t)tgt[row] * E_ + k]);
    }
}

// ---------------------------------------------------------------------------
// hh = h0 @ W_hh^T + b_ih + b_hh   (64 x 4096, K=1024)   [loop invariant]
// ---------------------------------------------------------------------------
__global__ void k_hh(const unsigned short* __restrict__ h0_bf,
                     const unsigned short* __restrict__ W_hh_bf,
                     const float* __restrict__ b_ih, const float* __restrict__ b_hh,
                     float* __restrict__ hh) {
    const int wave = (blockIdx.x * blockDim.x + threadIdx.x) >> 5;
    const int mt = wave & 3, nt = wave >> 2;
    const unsigned short* A  = h0_bf   + (size_t)(mt * 16) * H_;
    const unsigned short* Bp = W_hh_bf + (size_t)(nt * 16) * H_;
    v8f acc = {};
    for (int k0 = 0; k0 < H_; k0 += 32)
        acc = wmma_bf16(tile_bf(A, H_, k0), tile_bf(Bp, H_, k0), acc);
    const int lane = threadIdx.x & 31;
    const int n    = nt * 16 + (lane & 15);
    const int mrow = mt * 16 + ((lane >> 4) << 3);
    const float bias = b_ih[n] + b_hh[n];
#pragma unroll
    for (int r = 0; r < 8; ++r)
        hh[(size_t)(mrow + r) * NG + n] = acc[r] + bias;
}

// ---------------------------------------------------------------------------
// proj[b] = enc[b] @ W_att^T  (256 x 1024, K=2048) -> bf16. 65536 waves.
// ---------------------------------------------------------------------------
__global__ void k_proj(const unsigned short* __restrict__ enc_bf,
                       const unsigned short* __restrict__ W_att_bf,
                       unsigned short* __restrict__ proj_bf) {
    const int wave = (blockIdx.x * blockDim.x + threadIdx.x) >> 5;
    const int nt = wave & 63;
    const int mt = (wave >> 6) & 15;
    const int b  = wave >> 10;
    const unsigned short* A  = enc_bf   + ((size_t)b * S_ + mt * 16) * D2H;
    const unsigned short* Bp = W_att_bf + (size_t)(nt * 16) * D2H;
    v8f acc = {};
    for (int k0 = 0; k0 < D2H; k0 += 32) {
        if ((k0 & 255) == 0) __builtin_prefetch(Bp + (size_t)(threadIdx.x & 15) * D2H + k0 + 256, 0, 1);
        acc = wmma_bf16(tile_bf(A, D2H, k0), tile_bf(Bp, D2H, k0), acc);
    }
    const int lane = threadIdx.x & 31;
    const int n    = nt * 16 + (lane & 15);
    const int srow = mt * 16 + ((lane >> 4) << 3);
#pragma unroll
    for (int r = 0; r < 8; ++r)
        proj_bf[((size_t)b * S_ + srow + r) * H_ + n] = f2bf(acc[r]);
    if (__builtin_amdgcn_cluster_id_x() != 0)       // NOP unless clustered
        __builtin_amdgcn_s_cluster_barrier();
}

// ---------------------------------------------------------------------------
// gates = [y_t | o_prev] @ W_ih^T + hh  (64 x 4096, K=1536)
// Block = 8 waves = 4 m-tiles x 2 n-tiles; W_ih rows [32*blk, 32*blk+32) are
// staged K-chunk-wise (32 x 512 bf16 = 32 KB) into LDS by the TDM.
// ---------------------------------------------------------------------------
__global__ void k_gates(int t,
                        const unsigned short* __restrict__ y_bf,
                        const unsigned short* __restrict__ o_bf,
                        const unsigned short* __restrict__ W_ih_bf,
                        const float* __restrict__ hh,
                        float* __restrict__ gates) {
#ifdef USE_TDM
    __shared__ unsigned short shB[32 * KCHUNK];     // 32 KB
#endif
    const int w       = threadIdx.x >> 5;           // wave in block: 0..7
    const int mt      = w & 3;
    const int ntLocal = w >> 2;                     // 0..1
    const int nt      = blockIdx.x * 2 + ntLocal;   // 0..255
    const int lane = threadIdx.x & 31;
    const int r    = lane & 15;
    const int kofs = (lane >> 4) << 3;
    const int m    = mt * 16 + r;                   // batch row
    const unsigned short* yrow = y_bf + ((size_t)t * B_ + m) * E_;
    const unsigned short* orow = o_bf + (size_t)m * H_;   // zeroed before t==0
    v8f acc = {};
    for (int chunk = 0; chunk < KI / KCHUNK; ++chunk) {
        const int kbase = chunk * KCHUNK;
#ifdef USE_TDM
        if (w == 0) {                               // one wave issues the DMA
            const unsigned long long ga = (unsigned long long)(uintptr_t)
                (W_ih_bf + ((size_t)blockIdx.x * 32) * KI + kbase);
            const unsigned lds_off = (unsigned)(uintptr_t)(&shB[0]);
            u32x4 g0;
            g0.x = 1u;                                              // count=1
            g0.y = lds_off;                                         // lds_addr
            g0.z = (unsigned)(ga & 0xffffffffu);                    // global[31:0]
            g0.w = (unsigned)((ga >> 32) & 0x01ffffffu) | (2u << 30); // [56:32]|type=2
            i32x8 g1;
            g1.s0 = (int)(1u << 16);                                // data_size=2B
            g1.s1 = (int)(((unsigned)KI & 0xffffu) << 16);          // tensor_dim0 lo
            g1.s2 = (int)(((unsigned)NG & 0xffffu) << 16);          // dim0 hi=0|dim1 lo
            g1.s3 = (int)((unsigned)KCHUNK << 16);                  // dim1 hi=0|tile_dim0
            g1.s4 = 32;                                             // tile_dim1=32
            g1.s5 = KI;                                             // dim0_stride lo32
            g1.s6 = 0;
            g1.s7 = 0;
            i32x4 z4 = {0, 0, 0, 0};
            i32x8 z8 = {0, 0, 0, 0, 0, 0, 0, 0};
            __builtin_amdgcn_tensor_load_to_lds(g0, g1, z4, z4, z8, 0);
            __builtin_amdgcn_s_wait_tensorcnt(0);
        }
        __syncthreads();
#endif
        for (int k0 = kbase; k0 < kbase + KCHUNK; k0 += 32) {
            const int kk = k0 + kofs;
            TileU a;
#pragma unroll
            for (int half = 0; half < 2; ++half) {
                const int kb = kk + half * 16;      // 8-aligned, never crosses 512
                const unsigned short* p = (kb < E_) ? (yrow + kb) : (orow + (kb - E_));
                a.q[half] = *(const v4u*)p;
            }
#ifdef USE_TDM
            const v16bf bt = tile_bf(shB + (size_t)(ntLocal * 16) * KCHUNK,
                                     KCHUNK, k0 - kbase);
#else
            const v16bf bt = tile_bf(W_ih_bf + (size_t)(nt * 16) * KI, KI, k0);
#endif
            acc = wmma_bf16(a.b, bt, acc);
        }
#ifdef USE_TDM
        __syncthreads();                            // before next chunk overwrite
#endif
    }
    const int n    = nt * 16 + (lane & 15);
    const int mrow = mt * 16 + ((lane >> 4) << 3);
#pragma unroll
    for (int rr = 0; rr < 8; ++rr) {
        const size_t o = (size_t)(mrow + rr) * NG + n;
        gates[o] = acc[rr] + hh[o];
    }
}

// ---------------------------------------------------------------------------
// LSTM pointwise: h = sig(o)*tanh(sig(f)*c0 + sig(i)*tanh(g)); store bf16.
// ---------------------------------------------------------------------------
__global__ void k_point(const float* __restrict__ gates, const float* __restrict__ c0,
                        unsigned short* __restrict__ h_bf) {
    const int idx = blockIdx.x * blockDim.x + threadIdx.x;   // B*H
    const int b = idx >> 10, j = idx & (H_ - 1);
    const float* g = gates + (size_t)b * NG;
    const float i_ = g[j], f_ = g[j + H_], gg = g[j + 2 * H_], o_ = g[j + 3 * H_];
    const float c = sigf(f_) * c0[idx] + sigf(i_) * tanhf(gg);
    h_bf[idx] = f2bf(sigf(o_) * tanhf(c));
}

// ---------------------------------------------------------------------------
// Attention: e = proj[b].h, mask, softmax over S, a = sum_s alpha*enc[b,s,:].
// h staged into LDS via async global->LDS when available.
// ---------------------------------------------------------------------------
__global__ void k_attn(const unsigned short* __restrict__ proj_bf,
                       const unsigned short* __restrict__ h_bf,
                       const int* __restrict__ mask,
                       const float* __restrict__ enc,
                       unsigned short* __restrict__ a_bf) {
    __shared__ unsigned short sh_h[H_];
    __shared__ float red[S_];
    __shared__ float s_alpha[S_];
    const int b = blockIdx.x, tid = threadIdx.x;
#if __has_builtin(__builtin_amdgcn_global_load_async_to_lds_b64)
    {   // 256 threads x 8B = 2 KB
        v2i*     g = (v2i*)(uintptr_t)(h_bf + (size_t)b * H_ + 4 * tid);
        lds_v2i* l = (lds_v2i*)(unsigned)(uintptr_t)(&sh_h[4 * tid]);
        __builtin_amdgcn_global_load_async_to_lds_b64(g, l, 0, 0);
#if __has_builtin(__builtin_amdgcn_s_wait_asynccnt)
        __builtin_amdgcn_s_wait_asynccnt(0);
#else
        asm volatile("s_wait_asynccnt 0" ::: "memory");
#endif
    }
#else
    for (int i = tid; i < H_; i += S_) sh_h[i] = h_bf[b * H_ + i];
#endif
    __syncthreads();

    const unsigned short* pr = proj_bf + ((size_t)b * S_ + tid) * H_;
    float e = 0.f;
    for (int k = 0; k < H_; k += 2)
        e += bf2f(pr[k]) * bf2f(sh_h[k]) + bf2f(pr[k + 1]) * bf2f(sh_h[k + 1]);
    if (mask[b * S_ + tid] != 0) e = -3.0e38f;

    red[tid] = e; __syncthreads();
    for (int off = 128; off; off >>= 1) {
        if (tid < off) red[tid] = fmaxf(red[tid], red[tid + off]);
        __syncthreads();
    }
    const float mx = red[0]; __syncthreads();
    const float ex = __expf(e - mx);
    red[tid] = ex; __syncthreads();
    for (int off = 128; off; off >>= 1) {
        if (tid < off) red[tid] += red[tid + off];
        __syncthreads();
    }
    s_alpha[tid] = ex * (1.0f / red[0]);
    __syncthreads();

#pragma unroll
    for (int i = 0; i < D2H / S_; ++i) {            // 8 columns per thread
        const int d = tid + i * S_;
        const float* ep = enc + (size_t)b * S_ * D2H + d;
        float acc = 0.f;
        for (int s = 0; s < S_; ++s) acc += s_alpha[s] * ep[(size_t)s * D2H];
        a_bf[(size_t)b * D2H + d] = f2bf(acc);
    }
}

// ---------------------------------------------------------------------------
// O_t = tanh([h | a] @ W_comb^T)  (64 x 1024, K=3072) -> d_out[t] + o_bf
// ---------------------------------------------------------------------------
__global__ void k_comb(const unsigned short* __restrict__ h_bf,
                       const unsigned short* __restrict__ a_bf,
                       const unsigned short* __restrict__ W_comb_bf,
                       float* __restrict__ out_t,
                       unsigned short* __restrict__ o_bf) {
    const int wave = (blockIdx.x * blockDim.x + threadIdx.x) >> 5;
    const int mt = wave & 3, nt = wave >> 2;        // nt < 64
    const int lane = threadIdx.x & 31;
    const int r    = lane & 15;
    const int kofs = (lane >> 4) << 3;
    const int m    = mt * 16 + r;
    const unsigned short* hrow = h_bf + (size_t)m * H_;
    const unsigned short* arow = a_bf + (size_t)m * D2H;
    const unsigned short* Brow = W_comb_bf + (size_t)(nt * 16 + r) * KC;
    v8f acc = {};
    for (int k0 = 0; k0 < KC; k0 += 32) {
        const int kk = k0 + kofs;
        TileU a;
#pragma unroll
        for (int half = 0; half < 2; ++half) {
            const int kb = kk + half * 16;          // 8-aligned, never crosses 1024
            const unsigned short* p = (kb < H_) ? (hrow + kb) : (arow + (kb - H_));
            a.q[half] = *(const v4u*)p;
        }
        TileU bt;
        bt.q[0] = *(const v4u*)(Brow + kk);
        bt.q[1] = *(const v4u*)(Brow + kk + 16);
        acc = wmma_bf16(a.b, bt.b, acc);
    }
    const int n    = nt * 16 + (lane & 15);
    const int mrow = mt * 16 + ((lane >> 4) << 3);
#pragma unroll
    for (int rr = 0; rr < 8; ++rr) {
        const float v = tanhf(acc[rr]);
        out_t[(size_t)(mrow + rr) * H_ + n] = v;
        o_bf [(size_t)(mrow + rr) * H_ + n] = f2bf(v);
    }
}

// ---------------------------------------------------------------------------
extern "C" void kernel_launch(void* const* d_in, const int* in_sizes, int n_in,
                              void* d_out, int out_size, void* d_ws, size_t ws_size,
                              hipStream_t stream) {
    const int*   tgt    = (const int*)  d_in[0];
    const float* enc    = (const float*)d_in[1];
    const int*   mask   = (const int*)  d_in[2];
    const float* h0     = (const float*)d_in[3];
    const float* c0     = (const float*)d_in[4];
    const float* embed  = (const float*)d_in[5];
    const float* W_ih   = (const float*)d_in[6];
    const float* W_hh   = (const float*)d_in[7];
    const float* b_ih   = (const float*)d_in[8];
    const float* b_hh   = (const float*)d_in[9];
    const float* W_att  = (const float*)d_in[10];
    const float* W_comb = (const float*)d_in[11];
    float* out = (float*)d_out;
    (void)in_sizes; (void)n_in; (void)out_size; (void)ws_size;

    // Workspace carve (~135 MB, all 16B-aligned sizes)
    uint8_t* w = (uint8_t*)d_ws;
    unsigned short* enc_bf    = (unsigned short*)w; w += (size_t)B_ * S_ * D2H * 2;
    unsigned short* proj_bf   = (unsigned short*)w; w += (size_t)B_ * S_ * H_ * 2;
    unsigned short* W_ih_bf   = (unsigned short*)w; w += (size_t)NG * KI * 2;
    unsigned short* W_comb_bf = (unsigned short*)w; w += (size_t)H_ * KC * 2;
    unsigned short* W_att_bf  = (unsigned short*)w; w += (size_t)H_ * D2H * 2;
    unsigned short* W_hh_bf   = (unsigned short*)w; w += (size_t)NG * H_ * 2;
    unsigned short* h0_bf     = (unsigned short*)w; w += (size_t)B_ * H_ * 2;
    unsigned short* y_bf      = (unsigned short*)w; w += (size_t)TSTEPS * B_ * E_ * 2;
    float*          hh        = (float*)w;          w += (size_t)B_ * NG * 4;
    float*          gates     = (float*)w;          w += (size_t)B_ * NG * 4;
    unsigned short* h_bf      = (unsigned short*)w; w += (size_t)B_ * H_ * 2;
    unsigned short* a_bf      = (unsigned short*)w; w += (size_t)B_ * D2H * 2;
    unsigned short* o_bf      = (unsigned short*)w; w += (size_t)B_ * H_ * 2;

    // ---- one-time precompute ----
    k_cvt<<<4096, 256, 0, stream>>>(enc,    enc_bf,    B_ * S_ * D2H);
    k_cvt<<<2048, 256, 0, stream>>>(W_ih,   W_ih_bf,   NG * KI);
    k_cvt<<<1024, 256, 0, stream>>>(W_comb, W_comb_bf, H_ * KC);
    k_cvt<<<1024, 256, 0, stream>>>(W_att,  W_att_bf,  H_ * D2H);
    k_cvt<<<1024, 256, 0, stream>>>(W_hh,   W_hh_bf,   NG * H_);
    k_cvt<<<32,   256, 0, stream>>>(h0,     h0_bf,     B_ * H_);
    k_embed<<<2048, 256, 0, stream>>>(tgt, embed, y_bf);
    k_zero<<<32, 256, 0, stream>>>((v4u*)o_bf, (B_ * H_) / 8);
    k_hh  <<<128,  256, 0, stream>>>(h0_bf, W_hh_bf, b_ih, b_hh, hh);
    k_proj<<<8192, 256, 0, stream>>>(enc_bf, W_att_bf, proj_bf);

    // ---- 63 serial decode steps ----
    for (int t = 0; t < TSTEPS; ++t) {
        k_gates<<<128, 256, 0, stream>>>(t, y_bf, o_bf, W_ih_bf, hh, gates);
        k_point<<<(B_ * H_) / 256, 256, 0, stream>>>(gates, c0, h_bf);
        k_attn <<<B_, S_, 0, stream>>>(proj_bf, h_bf, mask, enc, a_bf);
        k_comb <<<32, 256, 0, stream>>>(h_bf, a_bf, W_comb_bf,
                                        out + (size_t)t * B_ * H_, o_bf);
    }
}
